// CompoundEncoder_88871463289182
// MI455X (gfx1250) — matrix operand order repaired and checked by reading, not compiled, not run on hardware
//
#include <hip/hip_runtime.h>
#include <hip/hip_bf16.h>

// ---------------------------------------------------------------------------
// GINE encoder for MI455X (gfx1250, wave32, WMMA + TDM).
//   - weights pre-converted ONCE per launch to bf16, K-transposed, K-padded
//   - node MLPs: 64x128 block tile; weight tile pulled into LDS by the
//     Tensor Data Mover (tensor_load_to_lds, TENSORcnt) while VALU stages
//     bf16 activations; 8 waves x 4 M-subtiles of v_wmma_f32_16x16x32_bf16
//   - edge message: wave-per-edge coalesced gather + global f32 atomics
//   - pool: atomic segment-sum, then small per-graph projection
// ---------------------------------------------------------------------------

typedef __attribute__((ext_vector_type(16))) __bf16        v16bf;
typedef __attribute__((ext_vector_type(8)))  __bf16        v8bf;
typedef __attribute__((ext_vector_type(8)))  float         v8f;
typedef __attribute__((ext_vector_type(4)))  unsigned int  v4u;
typedef __attribute__((ext_vector_type(8)))  int           v8i;
typedef __attribute__((ext_vector_type(4)))  int           v4i;

#define LN_EPS 1e-5f

// ---------------------------------------------------------------------------
// One-time weight prep: W (f32 [K,128]) -> Wt (bf16 [128][Kpad]), zero-padded.
// Makes each wave's B fragment a single contiguous 32B LDS load and makes the
// weight tile TDM-copyable as a flat 2D block.
// ---------------------------------------------------------------------------
__global__ __launch_bounds__(256) void conv_weight(
    const float* __restrict__ W, __bf16* __restrict__ Wt, int K, int Kpad) {
  const int i = blockIdx.x * 256 + threadIdx.x;
  if (i >= 128 * Kpad) return;
  const int n = i / Kpad, k = i % Kpad;
  Wt[i] = (__bf16)((k < K) ? W[k * 128 + n] : 0.f);
}

// ---------------------------------------------------------------------------
// Edge kernel: one wave (32 lanes) per edge.
//   msg = relu(x[src] + edge_attr[e] @ We + be);  agg[dst] += msg (atomics)
// ---------------------------------------------------------------------------
__global__ __launch_bounds__(256) void gine_edge_msg(
    const float* __restrict__ x, const int* __restrict__ eidx,
    const float* __restrict__ eattr, const float* __restrict__ We,
    const float* __restrict__ be, float* __restrict__ agg,
    int E, int D) {
  __shared__ float We_s[6 * 128];
  __shared__ float be_s[128];
  const int tid = threadIdx.x;
  for (int i = tid; i < 6 * D; i += 256) We_s[i] = We[i];
  for (int i = tid; i < D; i += 256) be_s[i] = be[i];
  __syncthreads();

  const int wave = tid >> 5, lane = tid & 31;
  const long e = (long)blockIdx.x * 8 + wave;
  if (e >= E) return;

  const int src = eidx[e];
  const int dst = eidx[(long)E + e];
  const float a0 = eattr[e * 6 + 0], a1 = eattr[e * 6 + 1],
              a2 = eattr[e * 6 + 2], a3 = eattr[e * 6 + 3],
              a4 = eattr[e * 6 + 4], a5 = eattr[e * 6 + 5];

  for (int d = lane; d < D; d += 32) {
    float acc = be_s[d];
    acc = fmaf(a0, We_s[0 * D + d], acc);
    acc = fmaf(a1, We_s[1 * D + d], acc);
    acc = fmaf(a2, We_s[2 * D + d], acc);
    acc = fmaf(a3, We_s[3 * D + d], acc);
    acc = fmaf(a4, We_s[4 * D + d], acc);
    acc = fmaf(a5, We_s[5 * D + d], acc);
    float m = x[(long)src * D + d] + acc;   // coalesced row gather per wave
    m = fmaxf(m, 0.f);
    atomicAdd(&agg[(long)dst * D + d], m);  // global_atomic_add_f32
  }
}

// ---------------------------------------------------------------------------
// Fused node MLP: Out = act(LN((X + Agg) @ W + bias) * gamma + beta)
// Block = 256 threads = 8 waves; block tile = 64 rows x 128 cols.
// Wave w owns cols [16w,16w+16) and 4 M-subtiles (B fragment reused 4x).
//
// Weight tile (bf16 [128][Kpad], contiguous) is DMA'd into LDS by the TDM:
// wave 0 issues tensor_load_to_lds with a 2D descriptor (ISA 8.3/8.4) and
// waits on TENSORcnt while the whole block stages bf16 activations.
//
// Fragment layouts used here (wave32; A/B contiguous k, same order in both):
//   A row-major [64][Kpad]: lane l -> m=l&15, half=l>>4;
//     elems = K kb+16*half..+15   (one 32B ds load)
//   B as Wt[n][k]: lane l -> n=n0+(l&15); elems = K kb+16*half..+15 (one 32B)
//   C/D: lane l -> col l&15, rows r+8*half.
// ---------------------------------------------------------------------------
__global__ __launch_bounds__(256) void gine_mlp_ln(
    const float* __restrict__ X, const float* __restrict__ Agg,
    const __bf16* __restrict__ Wt, const float* __restrict__ bias,
    const float* __restrict__ gamma, const float* __restrict__ beta,
    float* __restrict__ Out, int nRows, int K, int Kpad, int doRelu) {
  __shared__ __align__(32) __bf16 Bsub[128 * 128];  // Wt tile [n][k]
  __shared__ __align__(32) __bf16 Asub[64 * 128];   // activations [row][k]
  __shared__ float Ctile[64 * 128];
  __shared__ float s_mu[64], s_rs[64];

  const int tid = threadIdx.x;
  const long r0 = (long)blockIdx.x * 64;

  // ---- TDM: DMA the bf16 weight tile (128*Kpad elems, 2B each) into LDS ----
  if (tid < 32) {  // wave-uniform guard: exactly one TDM op per block
    const unsigned lds_addr = (unsigned)(uintptr_t)(&Bsub[0]);
    const unsigned long long ga = (unsigned long long)(uintptr_t)Wt;
    v4u g0;
    g0[0] = 1u;                                   // count=1, no gather
    g0[1] = lds_addr;                             // lds_addr [63:32]
    g0[2] = (unsigned)(ga & 0xffffffffu);         // global_addr [95:64]
    g0[3] = (unsigned)((ga >> 32) & 0x01ffffffu)  // global_addr [120:96]
            | (2u << 30);                         // type=2 ("image")
    v8i g1 = {};
    g1[0] = (1 << 16);                            // data_size=1 -> 2 bytes
    g1[1] = (int)(((unsigned)Kpad & 0xffffu) << 16);   // tensor_dim0 lo16
    g1[2] = (int)((((unsigned)Kpad >> 16) & 0xffffu)   // tensor_dim0 hi16
                  | (128u << 16));                     // tensor_dim1 lo16
    g1[3] = (int)(((unsigned)Kpad & 0xffffu) << 16);   // tile_dim0 = Kpad
    g1[4] = 128;                                  // tile_dim1=128, tile_dim2=0
    g1[5] = Kpad;                                 // tensor_dim0_stride lo32
    g1[6] = 0;
    g1[7] = 0;
    const v4i z4 = {};
    const v8i z8 = {};
    __builtin_amdgcn_tensor_load_to_lds(g0, g1, z4, z4, z8, 0);
    __builtin_amdgcn_s_wait_tensorcnt(0);         // drain TENSORcnt (wave 0)
  }

  // ---- stage activations (X + Agg) -> bf16 LDS, overlapped with the DMA ----
  for (int i = tid; i < 64 * Kpad; i += 256) {
    const int row = i / Kpad, k = i % Kpad;
    const long gr = r0 + row;
    float v = 0.f;
    if (gr < nRows && k < K) {
      v = X[gr * K + k];
      if (Agg) v += Agg[gr * K + k];
    }
    Asub[row * Kpad + k] = (__bf16)v;
  }
  __syncthreads();

  const int wave = tid >> 5, lane = tid & 31;
  const int half = lane >> 4, mn = lane & 15;
  const int n0 = wave * 16;

  v8f acc0 = {}, acc1 = {}, acc2 = {}, acc3 = {};
  for (int kb = 0; kb < Kpad; kb += 32) {
    const v16bf b = *(const v16bf*)&Bsub[(n0 + mn) * Kpad + kb + 16 * half];
    union { v16bf v; v8bf h[2]; } ua;
    const __bf16* abase = &Asub[mn * Kpad + kb + 16 * half];
    const int mstride = 16 * Kpad;

    ua.v = *(const v16bf*)(abase);
    acc0 = __builtin_amdgcn_wmma_f32_16x16x32_bf16(false, ua.v, false, b,
                                                   (short)0, acc0, false, false);
    ua.v = *(const v16bf*)(abase + mstride);
    acc1 = __builtin_amdgcn_wmma_f32_16x16x32_bf16(false, ua.v, false, b,
                                                   (short)0, acc1, false, false);
    ua.v = *(const v16bf*)(abase + 2 * mstride);
    acc2 = __builtin_amdgcn_wmma_f32_16x16x32_bf16(false, ua.v, false, b,
                                                   (short)0, acc2, false, false);
    ua.v = *(const v16bf*)(abase + 3 * mstride);
    acc3 = __builtin_amdgcn_wmma_f32_16x16x32_bf16(false, ua.v, false, b,
                                                   (short)0, acc3, false, false);
  }

  // ---- scatter f32 tiles + bias into LDS for cross-wave LayerNorm ----
  const int col = n0 + mn;
#pragma unroll
  for (int r = 0; r < 8; r++) {
    const int row = r + 8 * half;
    Ctile[(row) * 128 + col]      = acc0[r] + bias[col];
    Ctile[(row + 16) * 128 + col] = acc1[r] + bias[col];
    Ctile[(row + 32) * 128 + col] = acc2[r] + bias[col];
    Ctile[(row + 48) * 128 + col] = acc3[r] + bias[col];
  }
  __syncthreads();

  if (tid < 64) {
    float s = 0.f, s2 = 0.f;
    for (int j = 0; j < 128; j++) {
      const float v = Ctile[tid * 128 + j];
      s += v;
      s2 = fmaf(v, v, s2);
    }
    const float mu = s * (1.f / 128.f);
    const float var = fmaxf(s2 * (1.f / 128.f) - mu * mu, 0.f);
    s_mu[tid] = mu;
    s_rs[tid] = rsqrtf(var + LN_EPS);
  }
  __syncthreads();

  for (int i = tid; i < 64 * 128; i += 256) {
    const int row = i >> 7, cc = i & 127;
    const long gr = r0 + row;
    if (gr < nRows) {
      float v = (Ctile[i] - s_mu[row]) * s_rs[row] * gamma[cc] + beta[cc];
      if (doRelu) v = fmaxf(v, 0.f);
      Out[gr * 128 + cc] = v;
    }
  }
}

// ---------------------------------------------------------------------------
// Segment sum for global mean pool.
// ---------------------------------------------------------------------------
__global__ __launch_bounds__(256) void pool_accum(
    const float* __restrict__ h, const int* __restrict__ batch,
    float* __restrict__ sums, float* __restrict__ cnt, long total) {
  const long idx = (long)blockIdx.x * 256 + threadIdx.x;
  if (idx >= total) return;
  const int node = (int)(idx >> 7), d = (int)(idx & 127);
  const int g = batch[node];
  atomicAdd(&sums[(long)g * 128 + d], h[idx]);
  if (d == 0) atomicAdd(&cnt[g], 1.f);
}

// ---------------------------------------------------------------------------
// pooled = sums / max(cnt,1);  out = pooled @ Wp + bp.  One block per graph.
// ---------------------------------------------------------------------------
__global__ __launch_bounds__(128) void pool_project(
    const float* __restrict__ sums, const float* __restrict__ cnt,
    const float* __restrict__ Wp, const float* __restrict__ bp,
    float* __restrict__ out) {
  __shared__ float p[128];
  const int g = blockIdx.x, o = threadIdx.x;
  const float c = fmaxf(cnt[g], 1.f);
  p[o] = sums[(long)g * 128 + o] / c;
  __syncthreads();
  float acc = bp[o];
  for (int d = 0; d < 128; d++) acc = fmaf(p[d], Wp[d * 128 + o], acc);
  out[(long)g * 128 + o] = acc;
}

// ---------------------------------------------------------------------------
// Launch. Param order follows jax pytree flatten of setup_inputs():
//   0:x 1:edge_index 2:edge_attr 3:batch 4:Wp 5:bp
//   per layer (sorted keys): W1 W2 We b1 b2 be bt1 bt2 g1 g2
// ---------------------------------------------------------------------------
extern "C" void kernel_launch(void* const* d_in, const int* in_sizes, int n_in,
                              void* d_out, int out_size, void* d_ws,
                              size_t ws_size, hipStream_t stream) {
  const int NODE_DIM = 78;
  const int N = in_sizes[0] / NODE_DIM;  // 50000
  const int E = in_sizes[2] / 6;         // 800000
  const int G = out_size / 128;          // 1000

  const float* x     = (const float*)d_in[0];
  const int*   eidx  = (const int*)d_in[1];
  const float* eattr = (const float*)d_in[2];
  const int*   batch = (const int*)d_in[3];
  const float* Wp    = (const float*)d_in[4];
  const float* bp    = (const float*)d_in[5];

  const float* W1[2];  const float* W2[2];  const float* We[2];
  const float* b1[2];  const float* b2[2];  const float* be[2];
  const float* bt1[2]; const float* bt2[2]; const float* g1[2]; const float* g2[2];
  for (int l = 0; l < 2; l++) {
    const int base = 6 + l * 10;
    W1[l]  = (const float*)d_in[base + 0];
    W2[l]  = (const float*)d_in[base + 1];
    We[l]  = (const float*)d_in[base + 2];
    b1[l]  = (const float*)d_in[base + 3];
    b2[l]  = (const float*)d_in[base + 4];
    be[l]  = (const float*)d_in[base + 5];
    bt1[l] = (const float*)d_in[base + 6];
    bt2[l] = (const float*)d_in[base + 7];
    g1[l]  = (const float*)d_in[base + 8];
    g2[l]  = (const float*)d_in[base + 9];
  }

  // Workspace carve-up.
  char* wsb = (char*)d_ws;
  float* agg  = (float*)wsb;                       // N*128 f32
  float* hA   = agg  + (long)N * 128;              // N*128 f32
  float* hB   = hA   + (long)N * 128;              // N*128 f32
  float* sums = hB   + (long)N * 128;              // G*128 f32
  float* cnt  = sums + (long)G * 128;              // G f32
  // bf16 weight buffers (256B-aligned), [128][Kpad] each
  uintptr_t woff = (uintptr_t)(cnt + G);
  woff = (woff + 255) & ~(uintptr_t)255;
  __bf16* wt10 = (__bf16*)woff;                    // layer0 W1: 128*96
  __bf16* wt20 = wt10 + 128 * 96;                  // layer0 W2: 128*128
  __bf16* wt11 = wt20 + 128 * 128;                 // layer1 W1: 128*128
  __bf16* wt21 = wt11 + 128 * 128;                 // layer1 W2: 128*128
  (void)ws_size; (void)n_in;

  const dim3 blk256(256);
  const dim3 edgeGrid((E + 7) / 8);
  const dim3 mlpGrid((N + 63) / 64);

  // ---- one-time weight prep (bf16, transposed, padded) ----
  conv_weight<<<dim3((128 * 96 + 255) / 256), blk256, 0, stream>>>(W1[0], wt10, 78, 96);
  conv_weight<<<dim3((128 * 128 + 255) / 256), blk256, 0, stream>>>(W2[0], wt20, 128, 128);
  conv_weight<<<dim3((128 * 128 + 255) / 256), blk256, 0, stream>>>(W1[1], wt11, 128, 128);
  conv_weight<<<dim3((128 * 128 + 255) / 256), blk256, 0, stream>>>(W2[1], wt21, 128, 128);

  // ---------------- layer 0 (in_dim = 78) ----------------
  (void)hipMemsetAsync(agg, 0, (size_t)N * 128 * sizeof(float), stream);
  gine_edge_msg<<<edgeGrid, blk256, 0, stream>>>(x, eidx, eattr, We[0], be[0],
                                                 agg, E, NODE_DIM);
  gine_mlp_ln<<<mlpGrid, blk256, 0, stream>>>(x, agg, wt10, b1[0], g1[0],
                                              bt1[0], hA, N, NODE_DIM, 96, 1);
  gine_mlp_ln<<<mlpGrid, blk256, 0, stream>>>(hA, nullptr, wt20, b2[0], g2[0],
                                              bt2[0], hB, N, 128, 128, 1);

  // ---------------- layer 1 (in_dim = 128) ----------------
  (void)hipMemsetAsync(agg, 0, (size_t)N * 128 * sizeof(float), stream);
  gine_edge_msg<<<edgeGrid, blk256, 0, stream>>>(hB, eidx, eattr, We[1], be[1],
                                                 agg, E, 128);
  gine_mlp_ln<<<mlpGrid, blk256, 0, stream>>>(hB, agg, wt11, b1[1], g1[1],
                                              bt1[1], hA, N, 128, 128, 1);
  gine_mlp_ln<<<mlpGrid, blk256, 0, stream>>>(hA, nullptr, wt21, b2[1], g2[1],
                                              bt2[1], hB, N, 128, 128, 1);

  // ---------------- pool + projection ----------------
  (void)hipMemsetAsync(sums, 0, (size_t)G * 128 * sizeof(float), stream);
  (void)hipMemsetAsync(cnt, 0, (size_t)G * sizeof(float), stream);
  const long total = (long)N * 128;
  pool_accum<<<dim3((unsigned)((total + 255) / 256)), blk256, 0, stream>>>(
      hB, batch, sums, cnt, total);
  pool_project<<<dim3(G), dim3(128), 0, stream>>>(sums, cnt, Wp, bp,
                                                  (float*)d_out);
}
